// STHGCN_23287312679163
// MI455X (gfx1250) — compile-verified, matrix operand order checked by reading, not run on hardware
//
#include <hip/hip_runtime.h>
#include <hip/hip_bf16.h>

// ---------------- problem constants (match reference) ----------------
#define NN 150000
#define DD 128
#define HH 4
#define DHH 32
#define EE 300000
#define LL 4
#define AA 8
#define TIME_SCALE 24.0f
#define SLOPE 0.2f

// ---------------- WMMA types ----------------
typedef __attribute__((ext_vector_type(16))) __bf16 v16bf;
typedef __attribute__((ext_vector_type(8)))  float  v8f;

union BF16Frag { v16bf v; unsigned u[8]; };

__device__ __forceinline__ unsigned short f2bf(float f) {
    unsigned u = __float_as_uint(f);
    unsigned r = (u + 0x7FFFu + ((u >> 16) & 1u)) >> 16;   // RNE
    return (unsigned short)r;
}

// order-preserving float<->uint for atomicMax-based segment max
__device__ __forceinline__ unsigned fenc(float f) {
    unsigned u = __float_as_uint(f);
    return (u & 0x80000000u) ? ~u : (u | 0x80000000u);
}
__device__ __forceinline__ float fdec(unsigned u) {
    return (u & 0x80000000u) ? __uint_as_float(u ^ 0x80000000u)
                             : __uint_as_float(~u);
}

// ---------------- CDNA5 async global->LDS copy (ASYNCcnt path) ----------------
typedef __attribute__((address_space(3))) char lds_char_t;

__device__ __forceinline__ unsigned lds_offset_of(const void* generic_ptr) {
    return (unsigned)(size_t)(lds_char_t*)generic_ptr;   // generic -> AS(3) -> 32-bit offset
}
__device__ __forceinline__ void async_copy_b128(unsigned lds_byte_off, const void* gptr) {
    asm volatile("global_load_async_to_lds_b128 %0, %1, off"
                 :: "v"(lds_byte_off), "v"((unsigned long long)(size_t)gptr)
                 : "memory");
}
__device__ __forceinline__ void wait_asynccnt0() {
    asm volatile("s_wait_asynccnt 0x0" ::: "memory");
}

// ---------------- GEMM: C[rows,128] = A[rows,128] x B[128,128] ----------------
// bf16 WMMA 16x16x32, f32 accumulate. Block = 256 thr (8 wave32), 128-row slab.
// A slab staged f32 via global_load_async_to_lds_b128; B staged transposed bf16.
#define AP 132   // f32 LDS row stride (floats): 528B -> bank stride 4, conflict-free
#define KP 132   // bf16 LDS row stride (ushorts): 264B -> bank stride 2, conflict-free
#define GEMM_LDS_BYTES (128 * AP * 4 + 128 * KP * 2)   // 67584 + 33792 = 101376

__global__ __launch_bounds__(256)
void gemm128_bf16(const float* __restrict__ A, const float* __restrict__ B,
                  float* __restrict__ C, int rows) {
    extern __shared__ char smraw[];
    float*          ldsAf = (float*)smraw;                   // [128][AP] f32  (row, k)
    unsigned short* ldsB  = (unsigned short*)(smraw + 128 * AP * 4); // [128][KP] bf16 (n, k)

    const int tid = threadIdx.x;
    const int rowBase = blockIdx.x * 128;
    const unsigned ldsA_base = lds_offset_of(ldsAf);

    // stage A slab (128x128 f32) via async copies; clamp OOB rows (their C rows
    // are never stored, so clamped garbage is harmless)
    for (int it = 0; it < 16; ++it) {
        int idx = (it * 256 + tid) * 4;           // float index, 0..16383 step 4
        int r = idx >> 7, c = idx & 127;
        int rg = rowBase + r; if (rg > rows - 1) rg = rows - 1;
        async_copy_b128(ldsA_base + (unsigned)(r * AP + c) * 4u,
                        A + (size_t)rg * DD + c);
    }
    // stage B transposed: B[k][n] -> ldsB[n][k]  (overlaps with async A copies)
    for (int it = 0; it < 16; ++it) {
        int idx = (it * 256 + tid) * 4;
        int k = idx >> 7, n = idx & 127;
        float4 f = *(const float4*)(B + k * DD + n);
        ldsB[(n + 0) * KP + k] = f2bf(f.x);
        ldsB[(n + 1) * KP + k] = f2bf(f.y);
        ldsB[(n + 2) * KP + k] = f2bf(f.z);
        ldsB[(n + 3) * KP + k] = f2bf(f.w);
    }
    wait_asynccnt0();
    __syncthreads();

    const int wave = tid >> 5, lane = tid & 31;
    const int rloc = wave * 16 + (lane & 15);
    const int kup  = (lane & 16) >> 1;      // +8 for upper half-wave (ISA A/B layout)

    v8f acc[8];
    #pragma unroll
    for (int ct = 0; ct < 8; ++ct) acc[ct] = (v8f){0,0,0,0,0,0,0,0};

    for (int ks = 0; ks < 4; ++ks) {
        const int k0 = ks * 32 + kup;
        BF16Frag a;
        #pragma unroll
        for (int p = 0; p < 8; ++p) {
            int kk = k0 + ((p & 4) << 2) + ((p & 3) << 1);  // ISA 16-bit A layout
            float2 f = *(const float2*)&ldsAf[rloc * AP + kk];
            a.u[p] = (unsigned)f2bf(f.x) | ((unsigned)f2bf(f.y) << 16);
        }
        #pragma unroll
        for (int ct = 0; ct < 8; ++ct) {
            BF16Frag b;
            const int nloc = ct * 16 + (lane & 15);
            #pragma unroll
            for (int p = 0; p < 8; ++p) {
                int kk = k0 + ((p & 4) << 2) + ((p & 3) << 1);
                b.u[p] = *(const unsigned*)&ldsB[nloc * KP + kk];
            }
            acc[ct] = __builtin_amdgcn_wmma_f32_16x16x32_bf16(
                false, a.v, false, b.v, (short)0, acc[ct], false, false);
        }
    }

    // C layout: VGPR r -> M=r (lanes 0-15) / M=r+8 (lanes 16-31); N = lane&15
    const int rOutBase = rowBase + wave * 16 + ((lane & 16) >> 1);
    const int cBase = lane & 15;
    #pragma unroll
    for (int ct = 0; ct < 8; ++ct) {
        #pragma unroll
        for (int r = 0; r < 8; ++r) {
            int row = rOutBase + r;
            if (row < rows)
                C[(size_t)row * DD + ct * 16 + cBase] = acc[ct][r];
        }
    }
}

// ---------------- edge feature + kv_in ----------------
__global__ void edge_kv_kernel(const float* __restrict__ x,
                               const float* __restrict__ attr,
                               const float* __restrict__ dt,
                               const float* __restrict__ ds,
                               const int*   __restrict__ src,
                               const int*   __restrict__ et,
                               const float* __restrict__ Wa,
                               const float* __restrict__ time_w,
                               const float* __restrict__ time_b,
                               const float* __restrict__ dist_w,
                               const float* __restrict__ dist_b,
                               const float* __restrict__ type_table,
                               float* __restrict__ kv) {
    int gid = blockIdx.x * blockDim.x + threadIdx.x;
    int e = gid >> 7, d = gid & 127;
    if (e >= EE) return;
    int s  = src[e];
    int ty = et[e];
    float tdt = dt[e] * (1.0f / TIME_SCALE);
    float tds = ds[e];
    float acc = cosf(tdt * time_w[d] + time_b[d])
              + cosf(tds * dist_w[d] + dist_b[d])
              + type_table[ty * DD + d];
    #pragma unroll
    for (int a = 0; a < AA; ++a)
        acc += attr[(size_t)e * AA + a] * Wa[a * DD + d];
    kv[(size_t)e * DD + d] = x[(size_t)s * DD + d] + acc;
}

// ---------------- segment-softmax pipeline ----------------
__global__ void init_m_kernel(unsigned* __restrict__ m) {
    int i = blockIdx.x * blockDim.x + threadIdx.x;
    if (i < NN * HH) m[i] = fenc(-1e30f);
}

__global__ void logits_kernel(const float* __restrict__ q,
                              const float* __restrict__ kb,
                              const int* __restrict__ dst,
                              float* __restrict__ lg,
                              unsigned* __restrict__ m) {
    int gid = blockIdx.x * blockDim.x + threadIdx.x;
    if (gid >= EE * HH) return;
    int e = gid >> 2, h = gid & 3;
    int dn = dst[e];
    const float* qp = q  + (size_t)dn * DD + h * DHH;
    const float* kp = kb + (size_t)e  * DD + h * DHH;
    float dot = 0.f;
    #pragma unroll
    for (int j = 0; j < DHH; ++j) dot += qp[j] * kp[j];
    dot *= 0.17677669529663687f;                 // 32^-0.5
    dot = dot >= 0.f ? dot : SLOPE * dot;        // leaky relu
    lg[(size_t)e * HH + h] = dot;
    atomicMax(m + (size_t)dn * HH + h, fenc(dot));
}

__global__ void expsum_kernel(const int* __restrict__ dst,
                              const unsigned* __restrict__ m,
                              float* __restrict__ lg,
                              float* __restrict__ denom) {
    int gid = blockIdx.x * blockDim.x + threadIdx.x;
    if (gid >= EE * HH) return;
    int e = gid >> 2, h = gid & 3;
    int dn = dst[e];
    float a = expf(lg[(size_t)e * HH + h] - fdec(m[(size_t)dn * HH + h]));
    lg[(size_t)e * HH + h] = a;
    atomicAdd(denom + (size_t)dn * HH + h, a);
}

__global__ void scatter_kernel(const int* __restrict__ dst,
                               const float* __restrict__ lg,
                               const float* __restrict__ denom,
                               const float* __restrict__ vb,
                               float* __restrict__ agg) {
    int gid = blockIdx.x * blockDim.x + threadIdx.x;
    int e = gid >> 7, d = gid & 127;
    if (e >= EE) return;
    int h  = d >> 5;
    int dn = dst[e];
    float w = lg[(size_t)e * HH + h] / denom[(size_t)dn * HH + h];
    atomicAdd(agg + (size_t)dn * DD + d, w * vb[(size_t)e * DD + d]);
}

// ---------------- residual + BN stats ----------------
__global__ __launch_bounds__(128)
void out_stats_kernel(const float* __restrict__ skip,
                      const float* __restrict__ agg,
                      const float* __restrict__ beta_arr, int layer,
                      float* __restrict__ outb,
                      float* __restrict__ colsum, float* __restrict__ colsumsq) {
    int d  = threadIdx.x;
    int r0 = blockIdx.x * 128;
    float beta = beta_arr[layer];
    float s = 0.f, s2 = 0.f;
    for (int r = 0; r < 128; ++r) {
        int row = r0 + r;
        if (row >= NN) break;
        size_t i = (size_t)row * DD + d;
        float o = beta * skip[i] + (1.f - beta) * agg[i];
        outb[i] = o;
        s += o; s2 += o * o;
    }
    atomicAdd(colsum + d, s);
    atomicAdd(colsumsq + d, s2);
}

__global__ void finalize_kernel(const float* __restrict__ outb,
                                const float* __restrict__ colsum,
                                const float* __restrict__ colsumsq,
                                const float* __restrict__ gamma,
                                const float* __restrict__ bnb,
                                int layer, float* __restrict__ xout) {
    int gid = blockIdx.x * blockDim.x + threadIdx.x;
    int row = gid >> 7, d = gid & 127;
    if (row >= NN) return;
    const float invN = 1.0f / (float)NN;
    float mu  = colsum[d] * invN;
    float var = colsumsq[d] * invN - mu * mu;
    size_t i = (size_t)row * DD + d;
    float o = (outb[i] - mu) * rsqrtf(var + 1e-5f) * gamma[layer * DD + d]
            + bnb[layer * DD + d];
    xout[i] = fmaxf(o, 0.f);
}

// ---------------- host-side orchestration ----------------
extern "C" void kernel_launch(void* const* d_in, const int* in_sizes, int n_in,
                              void* d_out, int out_size, void* d_ws, size_t ws_size,
                              hipStream_t stream) {
    const float* x_in     = (const float*)d_in[0];
    const float* Wq       = (const float*)d_in[1];
    const float* Wk       = (const float*)d_in[2];
    const float* Wv       = (const float*)d_in[3];
    const float* Wskip    = (const float*)d_in[4];
    const float* Wa       = (const float*)d_in[5];
    const float* time_w   = (const float*)d_in[6];
    const float* time_b   = (const float*)d_in[7];
    const float* dist_w   = (const float*)d_in[8];
    const float* dist_b   = (const float*)d_in[9];
    const float* type_tab = (const float*)d_in[10];
    const float* bn_gamma = (const float*)d_in[11];
    const float* bn_beta  = (const float*)d_in[12];
    const float* beta_arr = (const float*)d_in[13];
    const float* e_attr   = (const float*)d_in[14];
    const float* e_dt     = (const float*)d_in[15];
    const float* e_ds     = (const float*)d_in[16];
    const int*   e_index  = (const int*)d_in[17];
    const int*   e_type   = (const int*)d_in[18];

    const size_t NBf = (size_t)NN * DD;   // 19.2M floats
    const size_t EBf = (size_t)EE * DD;   // 38.4M floats

    float* base = (float*)d_ws;
    float* xA    = base;
    float* xB    = xA + NBf;
    float* qb    = xB + NBf;
    float* sk    = qb + NBf;
    float* kv    = sk + NBf;
    float* kb    = kv + EBf;
    float* vb    = kb + EBf;
    float* agg   = vb + EBf;
    float* lg    = agg + NBf;                 // [E,H]
    float* denom = lg + (size_t)EE * HH;      // [N,H]
    float* csum  = denom + (size_t)NN * HH;   // [128]
    float* csq   = csum + DD;                 // [128]
    unsigned* mbuf = (unsigned*)(csq + DD);   // [N,H]

    float* pingpong[2] = { xA, xB };

    const int gemmGridN = (NN + 127) / 128;   // 1172
    const int gemmGridE = (EE + 127) / 128;   // 2344

    for (int l = 0; l < LL; ++l) {
        const float* xin  = (l == 0) ? x_in : pingpong[(l - 1) & 1];
        float*       xout = (l == LL - 1) ? (float*)d_out : pingpong[l & 1];

        const float* Wq_l = Wq    + (size_t)l * DD * DD;
        const float* Wk_l = Wk    + (size_t)l * DD * DD;
        const float* Wv_l = Wv    + (size_t)l * DD * DD;
        const float* Ws_l = Wskip + (size_t)l * DD * DD;
        const float* Wa_l = Wa    + (size_t)l * AA * DD;
        const float* at_l = e_attr + (size_t)l * EE * AA;
        const float* dt_l = e_dt   + (size_t)l * EE;
        const float* ds_l = e_ds   + (size_t)l * EE;
        const int*   src  = e_index + (size_t)l * 2 * EE;
        const int*   dst  = src + EE;
        const int*   et_l = e_type + (size_t)l * EE;

        // q = x@Wq ; skip = x@Wskip   (WMMA bf16, async-LDS staged)
        gemm128_bf16<<<gemmGridN, 256, GEMM_LDS_BYTES, stream>>>(xin, Wq_l, qb, NN);
        gemm128_bf16<<<gemmGridN, 256, GEMM_LDS_BYTES, stream>>>(xin, Ws_l, sk, NN);

        // kv_in = x[src] + edge_feat
        edge_kv_kernel<<<(int)((EBf + 255) / 256), 256, 0, stream>>>(
            xin, at_l, dt_l, ds_l, src, et_l, Wa_l,
            time_w, time_b, dist_w, dist_b, type_tab, kv);

        // k = kv@Wk ; v = kv@Wv   (WMMA bf16)
        gemm128_bf16<<<gemmGridE, 256, GEMM_LDS_BYTES, stream>>>(kv, Wk_l, kb, EE);
        gemm128_bf16<<<gemmGridE, 256, GEMM_LDS_BYTES, stream>>>(kv, Wv_l, vb, EE);

        // reset reduction buffers
        hipMemsetAsync(agg,   0, NBf * sizeof(float), stream);
        hipMemsetAsync(denom, 0, (size_t)NN * HH * sizeof(float), stream);
        hipMemsetAsync(csum,  0, 2 * DD * sizeof(float), stream);
        init_m_kernel<<<(NN * HH + 255) / 256, 256, 0, stream>>>(mbuf);

        // segment softmax + weighted scatter
        logits_kernel<<<(EE * HH + 255) / 256, 256, 0, stream>>>(qb, kb, dst, lg, mbuf);
        expsum_kernel<<<(EE * HH + 255) / 256, 256, 0, stream>>>(dst, mbuf, lg, denom);
        scatter_kernel<<<(int)((EBf + 255) / 256), 256, 0, stream>>>(dst, lg, denom, vb, agg);

        // residual + BN stats + normalize/ReLU  (qb is dead -> reuse as 'out')
        out_stats_kernel<<<(NN + 127) / 128, 128, 0, stream>>>(
            sk, agg, beta_arr, l, qb, csum, csq);
        finalize_kernel<<<(int)((NBf + 255) / 256), 256, 0, stream>>>(
            qb, csum, csq, bn_gamma, bn_beta, l, xout);
    }
}
